// naive_attn_like_55757265437272
// MI455X (gfx1250) — matrix operand order repaired and checked by reading, not compiled, not run on hardware
//
#include <hip/hip_runtime.h>
#include <hip/hip_bf16.h>

typedef __attribute__((ext_vector_type(16))) _Float16 v16h;
typedef __attribute__((ext_vector_type(8)))  _Float16 v8h;
typedef __attribute__((ext_vector_type(4)))  _Float16 v4h;
typedef __attribute__((ext_vector_type(8)))  float    v8f;

#define NTOK 8192
#define DIMK 1024
#define DKV  128

// ---------------------------------------------------------------------------
// Kernel 1: fused projection  qk = x@Wqk^T + bqk,  v = x@Wv^T + bv  (f16 out)
// Workgroup: 256 threads (8 waves). 64 rows of x per WG.
// wave w: row block = w%4 (16 rows), col half = w/4 (128 of 256 output cols).
// ---------------------------------------------------------------------------
__global__ __launch_bounds__(256) void proj_kernel(
    const float* __restrict__ x,    // [8192][1024]
    const float* __restrict__ Wqk,  // [128][1024]
    const float* __restrict__ bqk,  // [128]
    const float* __restrict__ Wv,   // [128][1024]
    const float* __restrict__ bv,   // [128]
    _Float16* __restrict__ qk_h,    // [8192][128]
    _Float16* __restrict__ v_h)     // [8192][128]
{
    __shared__ __align__(16) _Float16 xs[64][32];   // 4KB
    __shared__ __align__(16) _Float16 ws[256][32];  // 16KB

    const int tid  = threadIdx.x;
    const int lane = tid & 31;
    const int wave = tid >> 5;
    const int rowblk  = wave & 3;   // 0..3
    const int colhalf = wave >> 2;  // 0..1
    const int g = lane >> 4;        // half-wave group
    const int n = lane & 15;        // column within 16-wide tile
    const int wg_row0 = blockIdx.x * 64;

    v8f acc[8];
#pragma unroll
    for (int t = 0; t < 8; ++t)
#pragma unroll
        for (int r = 0; r < 8; ++r) acc[t][r] = 0.0f;

    for (int k0 = 0; k0 < DIMK; k0 += 32) {
        __syncthreads();
        // stage x tile 64x32 (f32 -> f16), vectorized float4 loads, coalesced
#pragma unroll
        for (int i = 0; i < 2; ++i) {
            int idx = (i * 256 + tid) * 4;    // 0..2044
            int r = idx >> 5, c = idx & 31;
            float4 f = *(const float4*)&x[(size_t)(wg_row0 + r) * DIMK + k0 + c];
            v4h h; h[0] = (_Float16)f.x; h[1] = (_Float16)f.y;
                   h[2] = (_Float16)f.z; h[3] = (_Float16)f.w;
            *(v4h*)&xs[r][c] = h;
        }
        // stage W tile 256x32 (rows 0..127 = Wqk, 128..255 = Wv), float4 loads
#pragma unroll
        for (int i = 0; i < 8; ++i) {
            int idx = (i * 256 + tid) * 4;    // 0..8188
            int r = idx >> 5, c = idx & 31;
            const float* Wsrc = (r < 128) ? (Wqk + (size_t)r * DIMK)
                                          : (Wv + (size_t)(r - 128) * DIMK);
            float4 f = *(const float4*)&Wsrc[k0 + c];
            v4h h; h[0] = (_Float16)f.x; h[1] = (_Float16)f.y;
                   h[2] = (_Float16)f.z; h[3] = (_Float16)f.w;
            *(v4h*)&ws[r][c] = h;
        }
        __syncthreads();

        // A operand: 16x32 chunk of x, lane L -> row n, K halves per ISA layout
        union { v16h v; v8h h[2]; } A;
        A.h[0] = *(const v8h*)&xs[rowblk * 16 + n][g * 8];
        A.h[1] = *(const v8h*)&xs[rowblk * 16 + n][16 + g * 8];

#pragma unroll
        for (int t = 0; t < 8; ++t) {
            // B operand: W^T chunk (32x16): B(K,N) = W[coltile+N][k0+K], K = g*8+h (h<8), 16+g*8+(h-8) (h>=8)
            union { v16h v; v8h h[2]; } B;
            const int wrow = colhalf * 128 + t * 16 + n;
            B.h[0] = *(const v8h*)&ws[wrow][g * 8];
            B.h[1] = *(const v8h*)&ws[wrow][16 + g * 8];
            acc[t] = __builtin_amdgcn_wmma_f32_16x16x32_f16(
                false, A.v, false, B.v, (short)0, acc[t], false, false);
        }
    }

    // epilogue: add bias, store f16.  C layout: VGPR r -> row r+8g, col n.
    const float*  bias = colhalf ? bv  : bqk;
    _Float16*     dst  = colhalf ? v_h : qk_h;
#pragma unroll
    for (int t = 0; t < 8; ++t) {
        int c128 = t * 16 + n;  // 0..127 within this half
#pragma unroll
        for (int r = 0; r < 8; ++r) {
            int row = wg_row0 + rowblk * 16 + r + 8 * g;
            float val = acc[t][r] + bias[c128];
            dst[(size_t)row * DKV + c128] = (_Float16)val;
        }
    }
}

// ---------------------------------------------------------------------------
// Kernel 2: flash attention.  score = qk@qk^T, softmax, out = P@v.
// Workgroup: 256 threads (8 waves); each wave owns 16 query rows.
// Per iteration: 32 keys staged in LDS. K tile staged with ASYNC global->LDS
// copies (no VGPR round-trip, ASYNCcnt-tracked); V tile transposed via VGPRs.
// ---------------------------------------------------------------------------
__global__ __launch_bounds__(256) void flash_kernel(
    const _Float16* __restrict__ qk,  // [8192][128]  (Q and K are identical)
    const _Float16* __restrict__ vh,  // [8192][128]
    float* __restrict__ out)          // [8192][128]
{
    __shared__ __align__(16) _Float16 Ks[32][128];     // 8KB  key tile
    __shared__ __align__(16) _Float16 Vt[128][32];     // 8KB  V tile, transposed [d][key]
    __shared__ __align__(16) float    Ss[8][16][32];   // 16KB per-wave S scratch
    __shared__ __align__(16) _Float16 Ps[8][16][32];   // 8KB  per-wave P scratch
    __shared__ float                  Al[8][16];       // per-row alpha / l

    const int tid  = threadIdx.x;
    const int lane = tid & 31;
    const int wave = tid >> 5;
    const int g = lane >> 4;
    const int n = lane & 15;
    const int q0 = blockIdx.x * 128 + wave * 16;  // this wave's query rows

    // Preload Q (16x128) in WMMA A-layout: 4 chunks of K=32
    v16h Aq[4];
#pragma unroll
    for (int c = 0; c < 4; ++c) {
        union { v16h v; v8h h[2]; } u;
        const _Float16* qrow = qk + (size_t)(q0 + n) * DKV + c * 32;
        u.h[0] = *(const v8h*)&qrow[g * 8];
        u.h[1] = *(const v8h*)&qrow[16 + g * 8];
        Aq[c] = u.v;
    }

    v8f O[8];
#pragma unroll
    for (int t = 0; t < 8; ++t)
#pragma unroll
        for (int r = 0; r < 8; ++r) O[t][r] = 0.0f;

    float m_i = -__builtin_inff();  // row max (valid in lanes 0..15, row = lane)
    float l_i = 0.0f;               // row sum

    // per-thread K-tile async-copy coordinates: 2 x 16B per thread = 8KB tile
    // element idx = (i*256 + tid)*8 halves; r = idx>>7 (key row), c = idx&127
    const int kr0 = (tid * 8) >> 7,          kc0 = (tid * 8) & 127;
    const int kr1 = ((256 + tid) * 8) >> 7,  kc1 = ((256 + tid) * 8) & 127;
    const unsigned lds_k0 = (unsigned)(uintptr_t)&Ks[kr0][kc0];
    const unsigned lds_k1 = (unsigned)(uintptr_t)&Ks[kr1][kc1];

    // per-thread prefetch coordinates: each thread covers 32B of a 32x256B tile
    const int pf_row = tid >> 3;         // 0..31
    const int pf_col = (tid & 7) * 16;   // halves

    for (int j0 = 0; j0 < NTOK; j0 += 32) {
        __syncthreads();
        // stage K tile 32x128 f16 via async global->LDS copies (ASYNCcnt)
        {
            unsigned long long ga0 =
                (unsigned long long)(uintptr_t)&qk[(size_t)(j0 + kr0) * DKV + kc0];
            unsigned long long ga1 =
                (unsigned long long)(uintptr_t)&qk[(size_t)(j0 + kr1) * DKV + kc1];
            asm volatile("global_load_async_to_lds_b128 %0, %1, off"
                         :: "v"(lds_k0), "v"(ga0) : "memory");
            asm volatile("global_load_async_to_lds_b128 %0, %1, off"
                         :: "v"(lds_k1), "v"(ga1) : "memory");
        }
        // stage V tile transposed: Vt[d][k] = v[j0+k][d]; vector global reads,
        // scattered LDS writes
#pragma unroll
        for (int i = 0; i < 2; ++i) {
            int idx = (i * 256 + tid) * 8;    // 0..4088
            int r = idx >> 7, c = idx & 127;
            v8h vv = *(const v8h*)&vh[(size_t)(j0 + r) * DKV + c];
#pragma unroll
            for (int k = 0; k < 8; ++k) Vt[c + k][r] = vv[k];
        }
        // prefetch next iteration's K/V tile into cache while we compute
        if (j0 + 32 < NTOK) {
            __builtin_prefetch(&qk[(size_t)(j0 + 32 + pf_row) * DKV + pf_col], 0, 3);
            __builtin_prefetch(&vh[(size_t)(j0 + 32 + pf_row) * DKV + pf_col], 0, 3);
        }
        // drain this wave's async copies before the data-ready barrier
        asm volatile("s_wait_asynccnt 0" ::: "memory");
        __syncthreads();

        // S tiles: S = Q @ K^T over dk=128 (4 chunks of K=32).
        // Two partial accumulators per tile -> 4 independent WMMA streams
        // (hides the 5-slot WMMA->WMMA RAW hazard).
        v8f S0a, S0b, S1a, S1b;
#pragma unroll
        for (int r = 0; r < 8; ++r) { S0a[r] = 0.0f; S0b[r] = 0.0f;
                                      S1a[r] = 0.0f; S1b[r] = 0.0f; }
#pragma unroll
        for (int c = 0; c < 2; ++c) {
            union { v16h v; v8h h[2]; } B0, B1, B2, B3;
            // B(K,N) = Ks[N(+16)][chunk*32 + K], K = g*8+h (h<8), 16+g*8+(h-8) (h>=8)
            B0.h[0] = *(const v8h*)&Ks[n][c * 32 + g * 8];
            B0.h[1] = *(const v8h*)&Ks[n][c * 32 + 16 + g * 8];
            B1.h[0] = *(const v8h*)&Ks[16 + n][c * 32 + g * 8];
            B1.h[1] = *(const v8h*)&Ks[16 + n][c * 32 + 16 + g * 8];
            B2.h[0] = *(const v8h*)&Ks[n][(c + 2) * 32 + g * 8];
            B2.h[1] = *(const v8h*)&Ks[n][(c + 2) * 32 + 16 + g * 8];
            B3.h[0] = *(const v8h*)&Ks[16 + n][(c + 2) * 32 + g * 8];
            B3.h[1] = *(const v8h*)&Ks[16 + n][(c + 2) * 32 + 16 + g * 8];
            S0a = __builtin_amdgcn_wmma_f32_16x16x32_f16(
                false, Aq[c], false, B0.v, (short)0, S0a, false, false);
            S1a = __builtin_amdgcn_wmma_f32_16x16x32_f16(
                false, Aq[c], false, B1.v, (short)0, S1a, false, false);
            S0b = __builtin_amdgcn_wmma_f32_16x16x32_f16(
                false, Aq[c + 2], false, B2.v, (short)0, S0b, false, false);
            S1b = __builtin_amdgcn_wmma_f32_16x16x32_f16(
                false, Aq[c + 2], false, B3.v, (short)0, S1b, false, false);
        }
        v8f S0, S1;
#pragma unroll
        for (int r = 0; r < 8; ++r) { S0[r] = S0a[r] + S0b[r];
                                      S1[r] = S1a[r] + S1b[r]; }

        // dump S (C layout: VGPR r -> row r+8g, col n / 16+n) to per-wave scratch
#pragma unroll
        for (int r = 0; r < 8; ++r) {
            Ss[wave][r + 8 * g][n]      = S0[r];
            Ss[wave][r + 8 * g][16 + n] = S1[r];
        }
        __syncthreads();

        // online softmax: lanes 0..15 each own one row of 32 scores
        if (lane < 16) {
            const int row = lane;
            float mx = m_i;
#pragma unroll
            for (int k = 0; k < 32; ++k) mx = fmaxf(mx, Ss[wave][row][k]);
            float alpha = __expf(m_i - mx);
            float sum = 0.0f;
#pragma unroll
            for (int k = 0; k < 32; ++k) {
                float p = __expf(Ss[wave][row][k] - mx);
                sum += p;
                Ps[wave][row][k] = (_Float16)p;
            }
            l_i = l_i * alpha + sum;
            m_i = mx;
            Al[wave][row] = alpha;
        }
        __syncthreads();

        // rescale O by per-row alpha
        float alr[8];
#pragma unroll
        for (int r = 0; r < 8; ++r) alr[r] = Al[wave][8 * g + r];
#pragma unroll
        for (int t = 0; t < 8; ++t)
#pragma unroll
            for (int r = 0; r < 8; ++r) O[t][r] *= alr[r];

        // P in A-layout (16x32), then O += P @ V for 8 column tiles
        // (8 independent accumulators -> no WMMA->WMMA hazard stalls here)
        union { v16h v; v8h h[2]; } P;
        P.h[0] = *(const v8h*)&Ps[wave][n][g * 8];
        P.h[1] = *(const v8h*)&Ps[wave][n][16 + g * 8];
#pragma unroll
        for (int t = 0; t < 8; ++t) {
            union { v16h v; v8h h[2]; } Bv;
            // B(K,N) = V[j0+K][t*16+N] = Vt[t*16+n][K], K = g*8+h (h<8), 16+g*8+(h-8) (h>=8)
            Bv.h[0] = *(const v8h*)&Vt[t * 16 + n][g * 8];
            Bv.h[1] = *(const v8h*)&Vt[t * 16 + n][16 + g * 8];
            O[t] = __builtin_amdgcn_wmma_f32_16x16x32_f16(
                false, P.v, false, Bv.v, (short)0, O[t], false, false);
        }
    }

    // broadcast row sums, normalize, store f32
    __syncthreads();
    if (lane < 16) Al[wave][lane] = l_i;
    __syncthreads();
    float lr[8];
#pragma unroll
    for (int r = 0; r < 8; ++r) lr[r] = Al[wave][8 * g + r];
#pragma unroll
    for (int t = 0; t < 8; ++t)
#pragma unroll
        for (int r = 0; r < 8; ++r)
            out[(size_t)(q0 + r + 8 * g) * DKV + t * 16 + n] = O[t][r] / lr[r];
}

// ---------------------------------------------------------------------------
extern "C" void kernel_launch(void* const* d_in, const int* in_sizes, int n_in,
                              void* d_out, int out_size, void* d_ws, size_t ws_size,
                              hipStream_t stream) {
    const float* x   = (const float*)d_in[0];
    const float* Wqk = (const float*)d_in[1];
    const float* bqk = (const float*)d_in[2];
    const float* Wv  = (const float*)d_in[3];
    const float* bv  = (const float*)d_in[4];
    float* out = (float*)d_out;

    _Float16* qk_h = (_Float16*)d_ws;                       // 8192*128 f16 = 2MB
    _Float16* v_h  = qk_h + (size_t)NTOK * DKV;             // +2MB

    proj_kernel<<<NTOK / 64, 256, 0, stream>>>(x, Wqk, bqk, Wv, bv, qk_h, v_h);
    flash_kernel<<<NTOK / 128, 256, 0, stream>>>(qk_h, v_h, out);
}
